// FFILinear_73023033966933
// MI455X (gfx1250) — compile-verified, matrix operand order checked by reading, not run on hardware
//
#include <hip/hip_runtime.h>

// ---------------------------------------------------------------------------
// Condensed sparse linear: y = x @ scatter(w, mask).T + b
// Strategy: scatter -> dense bf16 GEMM on v_wmma_f32_16x16x32_bf16 (CDNA5),
// double-buffered GLOBAL_LOAD_ASYNC_TO_LDS_B128 tile staging (ASYNCcnt path).
// ---------------------------------------------------------------------------

typedef __attribute__((ext_vector_type(16))) __bf16        v16bf;
typedef __attribute__((ext_vector_type(8)))  float         v8f;
typedef __attribute__((ext_vector_type(4)))  unsigned int  u32x4;

#define TOK  4096
#define INF  4096
#define OUTF 4096
#define NNZ  512

__device__ __forceinline__ unsigned short f32_to_bf16(float f) {
  unsigned int u = __float_as_uint(f);
  u += 0x7FFFu + ((u >> 16) & 1u);           // round-to-nearest-even
  return (unsigned short)(u >> 16);
}

// Kernel 1: zero the dense f32 weight scratch, convert x -> bf16.
__global__ void k_init(const float* __restrict__ x,
                       unsigned short* __restrict__ xbf,
                       float* __restrict__ wf) {
  unsigned i = blockIdx.x * 256u + threadIdx.x;   // exactly 4096*4096 threads
  xbf[i] = f32_to_bf16(x[i]);
  wf[i]  = 0.0f;
}

// Kernel 2: scatter condensed weights into dense W[o][in] (dup indices -> add).
__global__ void k_scatter(const float* __restrict__ w,
                          const int* __restrict__ mask,
                          float* __restrict__ wf) {
  unsigned i = blockIdx.x * 256u + threadIdx.x;   // OUTF*NNZ threads
  unsigned o = i >> 9;                            // / NNZ
  int idx = mask[i];
  atomicAdd(&wf[(size_t)o * INF + idx], w[i]);
}

// Kernel 3: dense W f32 -> bf16.
__global__ void k_cvt(const float* __restrict__ wf,
                      unsigned short* __restrict__ wbf) {
  unsigned i = blockIdx.x * 256u + threadIdx.x;
  wbf[i] = f32_to_bf16(wf[i]);
}

// Kernel 4: Y[T,O] = Xbf[T,K] * Wbf[O,K]^T + bias.
// Block = 256 threads = 8 waves; output tile 256(M) x 128(N); K step = 32.
// Wave grid 4(M) x 2(N): each wave owns 64x64 = 4x4 tiles of 16x16.
// Double-buffered async staging: issue tile k+1, compute tile k, then
// s_wait_asynccnt + one barrier per iteration.
#define BK      32
#define LDSPAD  40                 // 80B LDS row stride -> no 64B bank pattern
#define ASTRIDE (256 * LDSPAD)     // elements per A buffer
#define BSTRIDE (128 * LDSPAD)     // elements per B buffer
#define NKB     (INF / BK)         // 128 K steps

__device__ __forceinline__ void async_ld16(unsigned lds_addr, unsigned voff,
                                           const void* sbase) {
  asm volatile("global_load_async_to_lds_b128 %0, %1, %2"
               :: "v"(lds_addr), "v"(voff), "s"(sbase) : "memory");
}

__global__ void __launch_bounds__(256)
k_gemm(const unsigned short* __restrict__ A,   // Xbf [TOK][INF]
       const unsigned short* __restrict__ B,   // Wbf [OUTF][INF]
       const float* __restrict__ bias,
       float* __restrict__ Y) {
  __shared__ __align__(16) unsigned short As[2 * ASTRIDE];
  __shared__ __align__(16) unsigned short Bs[2 * BSTRIDE];

  const unsigned tid   = threadIdx.x;
  const unsigned wave  = tid >> 5;
  const unsigned lane  = tid & 31u;
  const unsigned laneM = lane & 15u;
  const unsigned khalf = (lane >> 4) * 8u;   // lanes 0-15: K base 0; 16-31: 8

  const unsigned t0 = blockIdx.y * 256u;
  const unsigned o0 = blockIdx.x * 128u;
  const unsigned wm = wave & 3u;             // 4 waves along M (64 rows each)
  const unsigned wn = wave >> 2;             // 2 waves along N (64 cols each)

  v8f acc[4][4] = {};

  // Async-copy descriptors: A tile = 256x32 -> 1024 16B chunks (4/thread),
  // B tile = 128x32 -> 512 chunks (2/thread). Chunk ch: row=ch/4, col=(ch%4)*8.
  unsigned aLds[4], aOff[4];     // LDS byte addr (buf 0), global byte offset
#pragma unroll
  for (int t = 0; t < 4; ++t) {
    unsigned ch  = tid + t * 256u;
    unsigned row = ch >> 2;
    unsigned col = (ch & 3u) * 8u;
    aLds[t] = (unsigned)(size_t)&As[row * LDSPAD + col];
    aOff[t] = ((t0 + row) * INF + col) * 2u;
  }
  unsigned bLds[2], bOff[2];
#pragma unroll
  for (int t = 0; t < 2; ++t) {
    unsigned ch  = tid + t * 256u;
    unsigned row = ch >> 2;
    unsigned col = (ch & 3u) * 8u;
    bLds[t] = (unsigned)(size_t)&Bs[row * LDSPAD + col];
    bOff[t] = ((o0 + row) * INF + col) * 2u;
  }

  auto issue_tile = [&](unsigned kb, unsigned buf) {
    const unsigned kbyte = kb * (BK * 2u);          // 64 bytes per K step
    const unsigned aboff = buf * (ASTRIDE * 2u);    // buffer byte offset in LDS
    const unsigned bboff = buf * (BSTRIDE * 2u);
#pragma unroll
    for (int t = 0; t < 4; ++t) async_ld16(aLds[t] + aboff, aOff[t] + kbyte, A);
#pragma unroll
    for (int t = 0; t < 2; ++t) async_ld16(bLds[t] + bboff, bOff[t] + kbyte, B);
  };

  union Frag { u32x4 u[2]; v16bf v; };

  auto compute_tile = [&](unsigned buf) {
    const unsigned short* Ab = As + buf * ASTRIDE;
    const unsigned short* Bb = Bs + buf * BSTRIDE;
    // A fragments: lane holds row M=laneM, K = {khalf..+7, khalf+16..+23}
    Frag af[4], bfr[4];
#pragma unroll
    for (int mi = 0; mi < 4; ++mi) {
      unsigned r = wm * 64u + mi * 16u + laneM;
      af[mi].u[0] = *(const u32x4*)(Ab + r * LDSPAD + khalf);
      af[mi].u[1] = *(const u32x4*)(Ab + r * LDSPAD + khalf + 16u);
    }
    // B fragments: lane holds column N=laneM, same K striping; Bb is [n][k].
#pragma unroll
    for (int ni = 0; ni < 4; ++ni) {
      unsigned cN = wn * 64u + ni * 16u + laneM;
      bfr[ni].u[0] = *(const u32x4*)(Bb + cN * LDSPAD + khalf);
      bfr[ni].u[1] = *(const u32x4*)(Bb + cN * LDSPAD + khalf + 16u);
    }
#pragma unroll
    for (int mi = 0; mi < 4; ++mi)
#pragma unroll
      for (int ni = 0; ni < 4; ++ni)
        acc[mi][ni] = __builtin_amdgcn_wmma_f32_16x16x32_bf16(
            false, af[mi].v, false, bfr[ni].v,
            (short)0, acc[mi][ni], false, false);
  };

  // Prologue: fetch tile 0 into buffer 0.
  issue_tile(0, 0);
  asm volatile("s_wait_asynccnt 0x0" ::: "memory");
  __syncthreads();

  // Ping-pong main loop: one barrier per iteration; next tile's async loads
  // fly behind the 16 WMMAs of the current tile.
  for (unsigned kb = 0; kb < NKB - 1; ++kb) {
    const unsigned cur = kb & 1u;
    issue_tile(kb + 1, cur ^ 1u);
    compute_tile(cur);
    asm volatile("s_wait_asynccnt 0x0" ::: "memory");
    __syncthreads();
  }
  compute_tile((NKB - 1) & 1u);

  // Epilogue: C/D layout — lanes 0-15: M=v; lanes 16-31: M=v+8; N=lane%16.
  const unsigned rofs = (lane >> 4) * 8u;
#pragma unroll
  for (int ni = 0; ni < 4; ++ni) {
    unsigned col = o0 + wn * 64u + ni * 16u + laneM;
    float bv = bias[col];
#pragma unroll
    for (int mi = 0; mi < 4; ++mi) {
      unsigned rbase = t0 + wm * 64u + mi * 16u + rofs;
#pragma unroll
      for (int v = 0; v < 8; ++v)
        Y[(size_t)(rbase + v) * OUTF + col] = acc[mi][ni][v] + bv;
    }
  }
}

extern "C" void kernel_launch(void* const* d_in, const int* in_sizes, int n_in,
                              void* d_out, int out_size, void* d_ws, size_t ws_size,
                              hipStream_t stream) {
  (void)in_sizes; (void)n_in; (void)out_size; (void)ws_size;

  const float* x    = (const float*)d_in[0];   // [TOK, INF] f32
  const float* w    = (const float*)d_in[1];   // [OUTF, NNZ] f32
  const int*   mask = (const int*)d_in[2];     // [OUTF, NNZ] i32
  const float* bias = (const float*)d_in[3];   // [OUTF] f32
  float*       y    = (float*)d_out;           // [TOK, OUTF] f32

  char* ws = (char*)d_ws;
  unsigned short* xbf = (unsigned short*)ws;                      // 32 MB
  unsigned short* wbf = (unsigned short*)(ws + (32ull << 20));    // 32 MB
  float*          wf  = (float*)(ws + (64ull << 20));             // 64 MB

  const unsigned NELEM = (unsigned)TOK * INF;                     // 16.7M

  k_init   <<<NELEM / 256, 256, 0, stream>>>(x, xbf, wf);
  k_scatter<<<(OUTF * NNZ) / 256, 256, 0, stream>>>(w, mask, wf);
  k_cvt    <<<NELEM / 256, 256, 0, stream>>>(wf, wbf);

  dim3 grid(OUTF / 128, TOK / 256);
  k_gemm<<<grid, 256, 0, stream>>>(xbf, wbf, bias, y);
}
